// GroupedQueryAttention_73555609911784
// MI455X (gfx1250) — matrix operand order repaired and checked by reading, not compiled, not run on hardware
//
#include <hip/hip_runtime.h>
#include <hip/hip_bf16.h>

// ---------------------------------------------------------------------------
// GQA forward for MI455X (gfx1250, wave32, WMMA).
// All matmuls on v_wmma_f32_16x16x32_bf16 (fp32 accumulate).
// GEMM: 32x64 tile/wave, ping-pong double buffering (no rotation copies).
// ---------------------------------------------------------------------------

typedef __attribute__((ext_vector_type(16))) __bf16 v16bf;
typedef __attribute__((ext_vector_type(8)))  __bf16 v8bf;
typedef __attribute__((ext_vector_type(8)))  float  v8f;

#define D_MODEL   2048
#define NUM_HEADS 16
#define NUM_KVH   4
#define HEAD_DIM  128
#define BATCH     2
#define T_SEQ     2048
#define MROWS     (BATCH * T_SEQ)       // 4096
#define D_KV      (NUM_KVH * HEAD_DIM)  // 512

// --- fragment helpers ------------------------------------------------------

__device__ __forceinline__ v16bf cat2(v8bf lo, v8bf hi) {
    v16bf r;
#pragma unroll
    for (int i = 0; i < 8; ++i) { r[i] = lo[i]; r[8 + i] = hi[i]; }
    return r;
}

// A-matrix 16x32 bf16 fragment (ISA 7.12.2): lane holds row = lane%16,
// K chunks [half*8 .. +7] and [16+half*8 .. +7]. base = row ptr + k0 (16B aligned).
__device__ __forceinline__ v16bf load_afrag(const __bf16* base, int half) {
    const __bf16* p = base + half * 8;
    return cat2(*(const v8bf*)p, *(const v8bf*)(p + 16));
}

// B-matrix 32x16 bf16 fragment: lane holds col N = lane%16, K chunks
// [half*8 .. +7] and [16+half*8 .. +7].  base = (row for this lane's N) ptr + k0, 32B aligned.
__device__ __forceinline__ v16bf load_bfrag(const __bf16* base, int half) {
    const __bf16* p = base + half * 8;
    return cat2(*(const v8bf*)p, *(const v8bf*)(p + 16));
}

__device__ __forceinline__ v8f wmma_bf16(v16bf a, v16bf b, v8f c) {
    return __builtin_amdgcn_wmma_f32_16x16x32_bf16(
        false, a, false, b, (short)0, c, false, false);
}

__device__ __forceinline__ float rowmax16(float v) {
#pragma unroll
    for (int m = 1; m < 16; m <<= 1) v = fmaxf(v, __shfl_xor(v, m, 32));
    return v;
}
__device__ __forceinline__ float rowsum16(float v) {
#pragma unroll
    for (int m = 1; m < 16; m <<= 1) v += __shfl_xor(v, m, 32);
    return v;
}

// --- conversion kernels ----------------------------------------------------

__global__ void __launch_bounds__(256) cvt_f32_bf16(const float* __restrict__ in,
                                                    __bf16* __restrict__ out,
                                                    int n) {
    int i = blockIdx.x * 256 + threadIdx.x;
    if (i < n) out[i] = (__bf16)in[i];
}

// W[K,N] fp32 row-major -> WT[N,K] bf16 row-major
__global__ void __launch_bounds__(256) cvt_transpose(const float* __restrict__ in,
                                                     __bf16* __restrict__ out,
                                                     int K, int N) {
    int idx = blockIdx.x * 256 + threadIdx.x;
    if (idx >= K * N) return;
    int n = idx / K;
    int k = idx - n * K;
    out[idx] = (__bf16)in[(size_t)k * N + n];
}

// vh [B*T, D_KV] -> vt [B, KVH, HEAD_DIM, T]
__global__ void __launch_bounds__(256) transpose_v(const __bf16* __restrict__ vh,
                                                   __bf16* __restrict__ vt) {
    int idx = blockIdx.x * 256 + threadIdx.x;      // over B*KVH*HEAD_DIM*T
    int s    = idx & (T_SEQ - 1);
    int rest = idx >> 11;
    int d    = rest & (HEAD_DIM - 1);
    int g    = rest >> 7;                           // b*KVH + kv
    int b    = g >> 2;
    int kv   = g & 3;
    vt[idx] = vh[((size_t)(b * T_SEQ) + s) * D_KV + kv * HEAD_DIM + d];
}

// --- GEMM: out[M,N] = A[M,K](bf16) x BT[N,K](bf16)^T + bias ---------------
// One wave -> 32x64 output tile (two 16-row A tiles share each B set).
// Ping-pong double buffering with 2x k-unroll: no buffer-rotation copies.
// K fixed at 2048 for all call sites.

__global__ void __launch_bounds__(256)
gemm_bf16_wmma(const __bf16* __restrict__ A, const __bf16* __restrict__ BT,
               const float* __restrict__ bias,
               float* __restrict__ outF, __bf16* __restrict__ outH,
               int M, int N) {
    constexpr int K = 2048;
    int wave = (blockIdx.x * 256 + threadIdx.x) >> 5;
    int lane = threadIdx.x & 31;
    int half = lane >> 4;
    int lm   = lane & 15;

    int ngroups = N >> 6;                 // N/64
    int mt = wave / ngroups;              // 32-row tiles
    int ng = wave - mt * ngroups;
    if (mt >= (M >> 5)) return;

    const __bf16* arow0 = A + (size_t)(mt * 32 + lm) * K;
    const __bf16* arow1 = arow0 + (size_t)16 * K;
    const __bf16* brow[4];
#pragma unroll
    for (int n = 0; n < 4; ++n)
        brow[n] = BT + (size_t)(ng * 64 + n * 16 + lm) * K;

    v8f acc0[4] = {}, acc1[4] = {};

    // prologue: stage set A at k = 0
    v16bf a0A = load_afrag(arow0, half);
    v16bf a1A = load_afrag(arow1, half);
    v16bf bA[4];
#pragma unroll
    for (int n = 0; n < 4; ++n) bA[n] = load_bfrag(brow[n], half);

    v16bf a0B, a1B, bB[4];

    for (int k0 = 0; k0 < K - 64; k0 += 64) {
        // stage set B (k0+32) while computing with set A (k0)
        a0B = load_afrag(arow0 + k0 + 32, half);
        a1B = load_afrag(arow1 + k0 + 32, half);
#pragma unroll
        for (int n = 0; n < 4; ++n) bB[n] = load_bfrag(brow[n] + k0 + 32, half);
#pragma unroll
        for (int n = 0; n < 4; ++n) acc0[n] = wmma_bf16(a0A, bA[n], acc0[n]);
#pragma unroll
        for (int n = 0; n < 4; ++n) acc1[n] = wmma_bf16(a1A, bA[n], acc1[n]);

        // stage set A (k0+64) while computing with set B (k0+32)
        a0A = load_afrag(arow0 + k0 + 64, half);
        a1A = load_afrag(arow1 + k0 + 64, half);
#pragma unroll
        for (int n = 0; n < 4; ++n) bA[n] = load_bfrag(brow[n] + k0 + 64, half);
#pragma unroll
        for (int n = 0; n < 4; ++n) acc0[n] = wmma_bf16(a0B, bB[n], acc0[n]);
#pragma unroll
        for (int n = 0; n < 4; ++n) acc1[n] = wmma_bf16(a1B, bB[n], acc1[n]);
    }
    // tail: set A holds k = K-64; stage set B at K-32, then drain both
    a0B = load_afrag(arow0 + K - 32, half);
    a1B = load_afrag(arow1 + K - 32, half);
#pragma unroll
    for (int n = 0; n < 4; ++n) bB[n] = load_bfrag(brow[n] + K - 32, half);
#pragma unroll
    for (int n = 0; n < 4; ++n) acc0[n] = wmma_bf16(a0A, bA[n], acc0[n]);
#pragma unroll
    for (int n = 0; n < 4; ++n) acc1[n] = wmma_bf16(a1A, bA[n], acc1[n]);
#pragma unroll
    for (int n = 0; n < 4; ++n) acc0[n] = wmma_bf16(a0B, bB[n], acc0[n]);
#pragma unroll
    for (int n = 0; n < 4; ++n) acc1[n] = wmma_bf16(a1B, bB[n], acc1[n]);

    // epilogue: C layout: VGPR r -> row r + 8*half, col = lane%16
#pragma unroll
    for (int n = 0; n < 4; ++n) {
        int col = ng * 64 + n * 16 + lm;
        float bb = bias ? bias[col] : 0.0f;
#pragma unroll
        for (int r = 0; r < 8; ++r) {
            int m0 = mt * 32 + r + 8 * half;
            int m1 = m0 + 16;
            float v0 = acc0[n][r] + bb;
            float v1 = acc1[n][r] + bb;
            if (outF) {
                outF[(size_t)m0 * N + col] = v0;
                outF[(size_t)m1 * N + col] = v1;
            }
            if (outH) {
                outH[(size_t)m0 * N + col] = (__bf16)v0;
                outH[(size_t)m1 * N + col] = (__bf16)v1;
            }
        }
    }
}

// --- Flash attention, causal, GQA ------------------------------------------
// One wave per (b, h, 16 query rows). K loop in steps of 32 keys.
// Loads are grouped ahead of the WMMA bursts; V fragments are issued before
// the softmax + LDS bounce so they overlap the VALU/trans/ds work.

__global__ void __launch_bounds__(256)
attn_wmma(const __bf16* __restrict__ qh, const __bf16* __restrict__ kh,
          const __bf16* __restrict__ vt, __bf16* __restrict__ aoh) {
    __shared__ __attribute__((aligned(32))) __bf16 lds[8][16 * 32];

    int wid  = threadIdx.x >> 5;
    int lane = threadIdx.x & 31;
    int half = lane >> 4;
    int lm   = lane & 15;

    int gw = blockIdx.x * 8 + wid;             // 0 .. B*H*(T/16)-1
    int tq = gw & ((T_SEQ / 16) - 1);
    int bh = gw >> 7;
    int h  = bh & (NUM_HEADS - 1);
    int b  = bh >> 4;
    int kv = h >> 2;

    const float scale = 0.08838834764831845f;  // 1/sqrt(128)

    // q A-fragments, K = 128 in 4 chunks of 32 (register resident)
    v16bf qa[4];
    const __bf16* qrow =
        qh + ((size_t)(b * T_SEQ) + tq * 16 + lm) * D_MODEL + h * HEAD_DIM;
#pragma unroll
    for (int c = 0; c < 4; ++c) qa[c] = load_afrag(qrow + c * 32, half);

    const __bf16* vbase = vt + (size_t)(b * NUM_KVH + kv) * HEAD_DIM * T_SEQ;

    v8f o[8] = {};
    float mrow[8], lrow[8];
#pragma unroll
    for (int r = 0; r < 8; ++r) { mrow[r] = -1e30f; lrow[r] = 0.0f; }

    const int tmax = tq * 16 + 15;
    for (int s0 = 0; s0 <= tmax; s0 += 32) {
        // ---- stage all 8 K fragments, then burst 8 score WMMAs ----
        const __bf16* krow0 =
            kh + ((size_t)(b * T_SEQ) + s0 + lm) * D_KV + kv * HEAD_DIM;
        const __bf16* krow1 = krow0 + (size_t)16 * D_KV;
        v16bf kb0[4], kb1[4];
#pragma unroll
        for (int c = 0; c < 4; ++c) {
            kb0[c] = load_bfrag(krow0 + c * 32, half);
            kb1[c] = load_bfrag(krow1 + c * 32, half);
        }
        v8f c0 = {}, c1 = {};
#pragma unroll
        for (int c = 0; c < 4; ++c) {
            c0 = wmma_bf16(qa[c], kb0[c], c0);
            c1 = wmma_bf16(qa[c], kb1[c], c1);
        }

        // ---- issue V fragment loads now; they overlap softmax + LDS ----
        v16bf vb[8];
#pragma unroll
        for (int n = 0; n < 8; ++n) {
            const __bf16* vp = vbase + (size_t)(n * 16 + lm) * T_SEQ + s0;
            vb[n] = load_bfrag(vp, half);
        }

        // ---- causal mask + online softmax ----
#pragma unroll
        for (int r = 0; r < 8; ++r) {
            int t  = tq * 16 + r + 8 * half;
            int sA = s0 + lm;
            int sB = s0 + 16 + lm;
            float x0 = (sA <= t) ? c0[r] * scale : -1e10f;
            float x1 = (sB <= t) ? c1[r] * scale : -1e10f;
            float mx = rowmax16(fmaxf(x0, x1));
            float mnew = fmaxf(mrow[r], mx);
            float al = __expf(mrow[r] - mnew);
            float p0 = __expf(x0 - mnew);
            float p1 = __expf(x1 - mnew);
            lrow[r] = lrow[r] * al + rowsum16(p0 + p1);
            mrow[r] = mnew;
#pragma unroll
            for (int n = 0; n < 8; ++n) o[n][r] *= al;
            // write P (16x32 tile) to this wave's LDS region, row = r+8*half
            lds[wid][(r + 8 * half) * 32 + lm]      = (__bf16)p0;
            lds[wid][(r + 8 * half) * 32 + 16 + lm] = (__bf16)p1;
        }
        asm volatile("s_wait_dscnt 0x0" ::: "memory");

        // ---- re-shape P: C layout -> A-fragment (16x32) ----
        const __bf16* pp = &lds[wid][lm * 32];
        v16bf pa = load_afrag(pp, half);

        // ---- O += P x V  (8 dtiles of 16, back-to-back WMMAs) ----
#pragma unroll
        for (int n = 0; n < 8; ++n) o[n] = wmma_bf16(pa, vb[n], o[n]);
    }

    // ---- normalize and store (bf16 for the O-projection) ----
    float inv[8];
#pragma unroll
    for (int r = 0; r < 8; ++r) inv[r] = 1.0f / lrow[r];
#pragma unroll
    for (int n = 0; n < 8; ++n)
#pragma unroll
        for (int r = 0; r < 8; ++r) {
            int t = tq * 16 + r + 8 * half;
            aoh[((size_t)(b * T_SEQ) + t) * D_MODEL + h * HEAD_DIM + n * 16 + lm] =
                (__bf16)(o[n][r] * inv[r]);
        }
}

// ---------------------------------------------------------------------------

extern "C" void kernel_launch(void* const* d_in, const int* in_sizes, int n_in,
                              void* d_out, int out_size, void* d_ws, size_t ws_size,
                              hipStream_t stream) {
    (void)in_sizes; (void)n_in; (void)out_size; (void)ws_size;
    const float* x  = (const float*)d_in[0];
    const float* Wq = (const float*)d_in[1];
    const float* bq = (const float*)d_in[2];
    const float* Wk = (const float*)d_in[3];
    const float* bk = (const float*)d_in[4];
    const float* Wv = (const float*)d_in[5];
    const float* bv = (const float*)d_in[6];
    const float* Wo = (const float*)d_in[7];
    const float* bo = (const float*)d_in[8];
    float* out = (float*)d_out;

    char* ws = (char*)d_ws;
    size_t off = 0;
    __bf16* xh  = (__bf16*)(ws + off); off += (size_t)MROWS * D_MODEL * 2;
    __bf16* WqT = (__bf16*)(ws + off); off += (size_t)D_MODEL * D_MODEL * 2;
    __bf16* WkT = (__bf16*)(ws + off); off += (size_t)D_KV * D_MODEL * 2;
    __bf16* WvT = (__bf16*)(ws + off); off += (size_t)D_KV * D_MODEL * 2;
    __bf16* WoT = (__bf16*)(ws + off); off += (size_t)D_MODEL * D_MODEL * 2;
    __bf16* qhb = (__bf16*)(ws + off); off += (size_t)MROWS * D_MODEL * 2;
    __bf16* khb = (__bf16*)(ws + off); off += (size_t)MROWS * D_KV * 2;
    __bf16* vhb = (__bf16*)(ws + off); off += (size_t)MROWS * D_KV * 2;
    __bf16* vtb = (__bf16*)(ws + off); off += (size_t)MROWS * D_KV * 2;
    __bf16* aoh = (__bf16*)(ws + off); off += (size_t)MROWS * D_MODEL * 2;

    // 1) convert activations & weights (weights transposed for B-fragments)
    cvt_f32_bf16<<<(MROWS * D_MODEL) / 256, 256, 0, stream>>>(x, xh, MROWS * D_MODEL);
    cvt_transpose<<<(D_MODEL * D_MODEL) / 256, 256, 0, stream>>>(Wq, WqT, D_MODEL, D_MODEL);
    cvt_transpose<<<(D_MODEL * D_KV) / 256, 256, 0, stream>>>(Wk, WkT, D_MODEL, D_KV);
    cvt_transpose<<<(D_MODEL * D_KV) / 256, 256, 0, stream>>>(Wv, WvT, D_MODEL, D_KV);
    cvt_transpose<<<(D_MODEL * D_MODEL) / 256, 256, 0, stream>>>(Wo, WoT, D_MODEL, D_MODEL);

    // 2) projections (WMMA GEMMs, bf16 outputs); 32-row tiles per wave
    {
        int blocksQ = (MROWS / 32) * (D_MODEL / 64) / 8;   // 512
        int blocksK = (MROWS / 32) * (D_KV / 64) / 8;      // 128
        gemm_bf16_wmma<<<blocksQ, 256, 0, stream>>>(xh, WqT, bq, nullptr, qhb,
                                                    MROWS, D_MODEL);
        gemm_bf16_wmma<<<blocksK, 256, 0, stream>>>(xh, WkT, bk, nullptr, khb,
                                                    MROWS, D_KV);
        gemm_bf16_wmma<<<blocksK, 256, 0, stream>>>(xh, WvT, bv, nullptr, vhb,
                                                    MROWS, D_KV);
    }

    // 3) v -> v^T per (b, kv-head) so P*V B-fragments are contiguous
    transpose_v<<<(MROWS * D_KV) / 256, 256, 0, stream>>>(vhb, vtb);

    // 4) flash attention (causal, GQA)
    {
        int waves = BATCH * NUM_HEADS * (T_SEQ / 16);      // 4096
        attn_wmma<<<waves / 8, 256, 0, stream>>>(qhb, khb, vtb, aoh);
    }

    // 5) output projection -> fp32 d_out
    {
        int blocks = (MROWS / 32) * (D_MODEL / 64) / 8;    // 512
        gemm_bf16_wmma<<<blocks, 256, 0, stream>>>(aoh, WoT, bo, out, nullptr,
                                                   MROWS, D_MODEL);
    }
}
